// ShapeAdaptiveAdapter_69234872811772
// MI455X (gfx1250) — compile-verified, matrix operand order checked by reading, not run on hardware
//
#include <hip/hip_runtime.h>
#include <hip/hip_bf16.h>

#define CCH   1024
#define CR    256
#define NADP  10
#define NANCH 3
#define HWSZ  4096
#define BATCH 2
#define NTOK  (BATCH * HWSZ)   // 8192
#define LN_EPS 1e-5f

#define BMT 128  // block tile M (tokens)
#define BNT 64   // block tile N (out columns)
#define BKT 64   // K chunk
#define LDT 72   // padded LDS row stride in bf16 elements (144B, 16B-aligned)

#ifndef SAA_USE_ASYNC
#define SAA_USE_ASYNC 1   // gfx1250 global_load_async_to_lds_* path
#endif

typedef __attribute__((ext_vector_type(16))) __bf16 v16bf;
typedef __attribute__((ext_vector_type(8)))  float  v8f;

__device__ __forceinline__ unsigned short f2bf(float f) {
    unsigned u = __float_as_uint(f);
    u += 0x7FFFu + ((u >> 16) & 1u);   // round to nearest even
    return (unsigned short)(u >> 16);
}

union Frag { v16bf v; uint4 u[2]; };

// 16-bit A/B fragment load per ISA layout: lanes 0-15 take K {0..7,16..23},
// lanes 16-31 take K {8..15,24..31} -> 16B chunks (laneHi) and (2+laneHi).
__device__ __forceinline__ v16bf load_frag_row(const unsigned short* row, int laneHi) {
    Frag f;
    const uint4* p = (const uint4*)row;
    f.u[0] = p[laneHi];
    f.u[1] = p[2 + laneHi];
    return f.v;
}

__device__ __forceinline__ v8f wmma_bf16(v16bf a, v16bf b, v8f c) {
    return __builtin_amdgcn_wmma_f32_16x16x32_bf16(false, a, false, b,
                                                   (short)0, c, false, false);
}

// One 16B global->LDS chunk. Async path: ASYNCcnt-tracked DMA, no VGPR bounce.
__device__ __forceinline__ void g2l_chunk(unsigned short* lgen, unsigned ldsOff,
                                          const unsigned short* gbase, int byteOff) {
#if SAA_USE_ASYNC
    asm volatile("global_load_async_to_lds_b128 %0, %1, %2"
                 :: "v"(ldsOff), "v"(byteOff), "s"(gbase) : "memory");
#else
    *(uint4*)lgen = *(const uint4*)((const char*)gbase + byteOff);
#endif
}

__device__ __forceinline__ void wait_g2l() {
#if SAA_USE_ASYNC
    asm volatile("s_wait_asynccnt 0x0" ::: "memory");
#endif
}

// ---------------- weight prep ----------------

// w1t[a][d][c] = ln_scale[a][c] * w1[a][c][d]   (bf16, K(=c)-contiguous)
__global__ void saa_prep_w1(const float* __restrict__ w1, const float* __restrict__ ls,
                            unsigned short* __restrict__ w1t) {
    const size_t n = (size_t)NADP * CCH * CR;
    for (size_t i = (size_t)blockIdx.x * blockDim.x + threadIdx.x; i < n;
         i += (size_t)gridDim.x * blockDim.x) {
        int d = (int)(i % CR);
        size_t r = i / CR;
        int c = (int)(r % CCH);
        int a = (int)(r / CCH);
        float v = ls[a * CCH + c] * w1[i];
        w1t[((size_t)a * CR + d) * CCH + c] = f2bf(v);
    }
}

// w2t[a][c][d] = w2[a][d][c]   (bf16, K(=d)-contiguous)
__global__ void saa_prep_w2(const float* __restrict__ w2, unsigned short* __restrict__ w2t) {
    const size_t n = (size_t)NADP * CR * CCH;
    for (size_t i = (size_t)blockIdx.x * blockDim.x + threadIdx.x; i < n;
         i += (size_t)gridDim.x * blockDim.x) {
        int c = (int)(i % CCH);
        size_t r = i / CCH;
        int d = (int)(r % CR);
        int a = (int)(r / CR);
        w2t[((size_t)a * CCH + c) * CR + d] = f2bf(w2[i]);
    }
}

// hbias[a][d] = sum_c ln_bias[a][c] * w1[a][c][d]
__global__ void saa_prep_hbias(const float* __restrict__ w1, const float* __restrict__ lb,
                               float* __restrict__ hb) {
    const int a = blockIdx.x, d = threadIdx.x;
    float s = 0.f;
    for (int c = 0; c < CCH; ++c)
        s += lb[a * CCH + c] * w1[((size_t)a * CCH + c) * CR + d];
    hb[a * CR + d] = s;
}

// ---------------- layernorm (NCHW -> token-major bf16) ----------------

__global__ __launch_bounds__(256) void saa_ln(const float* __restrict__ feat,
                                              unsigned short* __restrict__ xn) {
    extern __shared__ float tile[];            // [1024][33] floats (dynamic, 132KB)
    __shared__ float red[8][33], red2[8][33], muS[32], rsS[32];
    const int tid = threadIdx.x;
    const int tx = tid & 31, ty = tid >> 5;
    const int tok0 = blockIdx.x * 32;
    const int b = tok0 >> 12;
    const int hw0 = tok0 & (HWSZ - 1);
    const float* src = feat + (size_t)b * CCH * HWSZ + hw0 + tx;

    float s = 0.f, s2 = 0.f;
    for (int c = ty; c < CCH; c += 8) {
        float v = src[(size_t)c * HWSZ];
        tile[c * 33 + tx] = v;
        s += v; s2 += v * v;
    }
    red[ty][tx] = s; red2[ty][tx] = s2;
    __syncthreads();
    if (ty == 0) {
        float a = 0.f, a2 = 0.f;
        for (int j = 0; j < 8; ++j) { a += red[j][tx]; a2 += red2[j][tx]; }
        float mu = a * (1.0f / CCH);
        float var = a2 * (1.0f / CCH) - mu * mu;
        muS[tx] = mu;
        rsS[tx] = rsqrtf(var + LN_EPS);
    }
    __syncthreads();
    for (int t = 0; t < 4; ++t) {
        int tk = ty * 4 + t;
        float mu = muS[tk], rs = rsS[tk];
        unsigned short* dst = xn + (size_t)(tok0 + tk) * CCH;
        for (int j = 0; j < 32; ++j) {
            int c = tx + 32 * j;
            dst[c] = f2bf((tile[c * 33 + tk] - mu) * rs);
        }
    }
}

// ---------------- per-token adapter weights ----------------

__global__ void saa_cnt(const float* __restrict__ ar, float* __restrict__ cnt) {
    int tok = blockIdx.x * blockDim.x + threadIdx.x;
    if (tok >= NTOK) return;
    const int b = tok >> 12, hw = tok & (HWSZ - 1);
    float c[NADP];
#pragma unroll
    for (int a = 0; a < NADP; ++a) c[a] = 0.f;
    const float thr[9] = {1.f/9.f, 1.f/7.f, 1.f/5.f, 1.f/3.f, 1.f, 3.f, 5.f, 7.f, 9.f};
    for (int na = 0; na < NANCH; ++na) {
        float v = ar[((size_t)b * NANCH + na) * HWSZ + hw];
        int idx = 0;
#pragma unroll
        for (int j = 0; j < 9; ++j) idx += (v > thr[j]) ? 1 : 0;
        c[idx] += (1.0f / NANCH);
    }
#pragma unroll
    for (int a = 0; a < NADP; ++a) cnt[(size_t)tok * NADP + a] = c[a];
}

// ---------------- GEMM1: h[a] = gelu(xn @ w1t[a]^T + hbias[a]) ----------------
// 128x64 block tile, 8 waves each computing 32x32 (2x2 WMMA tiles),
// async double-buffered LDS.

__global__ __launch_bounds__(256) void saa_gemm1(
    const unsigned short* __restrict__ xn,   // [N][1024] bf16
    const unsigned short* __restrict__ w1t,  // [A][256][1024] bf16
    const float* __restrict__ hb,            // [A][256]
    unsigned short* __restrict__ h)          // [A][N][256] bf16
{
    __shared__ __align__(16) unsigned short smem[2 * (BMT + BNT) * LDT]; // 54KB
    const int a = blockIdx.z;
    const int t0 = blockIdx.x * BMT;
    const int d0 = blockIdx.y * BNT;
    const int tid = threadIdx.x;
    const int lane = tid & 31, wv = tid >> 5;
    const int wm = (wv & 3) * 32, wn = (wv >> 2) * 32;
    const int laneHi = lane >> 4, laneLo = lane & 15;
    const unsigned ldsBase = (unsigned)(size_t)smem;

    const unsigned short* A0 = xn + (size_t)t0 * CCH;
    const unsigned short* B0 = w1t + ((size_t)a * CR + d0) * CCH;

    v8f acc[2][2] = {};

    auto issue = [&](int k0, int buf) {
        const int bufEl = buf * (BMT + BNT) * LDT;
        unsigned short* xg = smem + bufEl;
        unsigned short* wg = xg + BMT * LDT;
        const unsigned xb = ldsBase + (unsigned)(bufEl * 2);
        const unsigned wb = xb + (unsigned)(BMT * LDT * 2);
#pragma unroll
        for (int r = 0; r < 4; ++r) {               // 1024 X chunks / 256 thr
            int ch = tid + r * 256;
            int row = ch >> 3, cc = ch & 7;
            int eo = row * LDT + cc * 8;
            g2l_chunk(xg + eo, xb + (unsigned)(eo * 2), A0,
                      (row * CCH + k0 + cc * 8) * 2);
        }
#pragma unroll
        for (int r = 0; r < 2; ++r) {               // 512 W chunks / 256 thr
            int ch = tid + r * 256;
            int row = ch >> 3, cc = ch & 7;
            int eo = row * LDT + cc * 8;
            g2l_chunk(wg + eo, wb + (unsigned)(eo * 2), B0,
                      (row * CCH + k0 + cc * 8) * 2);
        }
    };
    auto compute = [&](int buf) {
        const unsigned short* Xb = smem + buf * (BMT + BNT) * LDT;
        const unsigned short* Wb = Xb + BMT * LDT;
#pragma unroll
        for (int ks = 0; ks < BKT; ks += 32) {
            v16bf a0f = load_frag_row(&Xb[(wm + laneLo) * LDT + ks], laneHi);
            v16bf a1f = load_frag_row(&Xb[(wm + 16 + laneLo) * LDT + ks], laneHi);
            v16bf b0f = load_frag_row(&Wb[(wn + laneLo) * LDT + ks], laneHi);
            v16bf b1f = load_frag_row(&Wb[(wn + 16 + laneLo) * LDT + ks], laneHi);
            acc[0][0] = wmma_bf16(a0f, b0f, acc[0][0]);
            acc[0][1] = wmma_bf16(a0f, b1f, acc[0][1]);
            acc[1][0] = wmma_bf16(a1f, b0f, acc[1][0]);
            acc[1][1] = wmma_bf16(a1f, b1f, acc[1][1]);
        }
    };

    issue(0, 0);
    wait_g2l();
    __syncthreads();
    int buf = 0;
    for (int k0 = BKT; k0 < CCH; k0 += BKT) {
        issue(k0, buf ^ 1);     // prefetch next tile (other buffer)
        compute(buf);           // WMMAs overlap the async DMA
        wait_g2l();
        __syncthreads();
        buf ^= 1;
    }
    compute(buf);

    unsigned short* hout = h + (size_t)a * NTOK * CR;
#pragma unroll
    for (int tj = 0; tj < 2; ++tj) {
        const int d = d0 + wn + tj * 16 + laneLo;
        const float bias = hb[a * CR + d];
#pragma unroll
        for (int ti = 0; ti < 2; ++ti) {
#pragma unroll
            for (int i = 0; i < 8; ++i) {
                int tok = t0 + wm + ti * 16 + laneHi * 8 + i;
                float x = acc[ti][tj][i] + bias;
                float g = 0.5f * x * (1.0f + erff(x * 0.70710678118f));
                hout[(size_t)tok * CR + d] = f2bf(g);
            }
        }
    }
}

// ---------------- GEMM2: out = sum_a cnt[a,tok]*(h[a] @ w2t[a]^T) + feat ----------------
// Same tiling; 40 pipeline stages (adapter x k-chunk) so prefetch crosses
// adapter boundaries; per-adapter result folded into total with cnt weight.

__global__ __launch_bounds__(256) void saa_gemm2(
    const unsigned short* __restrict__ h,    // [A][N][256] bf16
    const unsigned short* __restrict__ w2t,  // [A][1024][256] bf16
    const float* __restrict__ cnt,           // [N][10]
    const float* __restrict__ feat,          // [B][C][HW]
    float* __restrict__ out)                 // [B][C][HW]
{
    __shared__ __align__(16) unsigned short smem[2 * (BMT + BNT) * LDT];
    __shared__ float cntS[BMT][NADP];
    const int t0 = blockIdx.x * BMT;
    const int c0 = blockIdx.y * BNT;
    const int tid = threadIdx.x;
    const int lane = tid & 31, wv = tid >> 5;
    const int wm = (wv & 3) * 32, wn = (wv >> 2) * 32;
    const int laneHi = lane >> 4, laneLo = lane & 15;
    const unsigned ldsBase = (unsigned)(size_t)smem;

    for (int i = tid; i < BMT * NADP; i += 256)
        cntS[i / NADP][i % NADP] = cnt[(size_t)(t0 + i / NADP) * NADP + i % NADP];

    v8f tot[2][2] = {};
    v8f acc[2][2] = {};

    auto issue = [&](int s, int buf) {
        const int aa = s >> 2, k0 = (s & 3) * BKT;
        const unsigned short* A0 = h + ((size_t)aa * NTOK + t0) * CR;
        const unsigned short* B0 = w2t + ((size_t)aa * CCH + c0) * CR;
        const int bufEl = buf * (BMT + BNT) * LDT;
        unsigned short* xg = smem + bufEl;
        unsigned short* wg = xg + BMT * LDT;
        const unsigned xb = ldsBase + (unsigned)(bufEl * 2);
        const unsigned wb = xb + (unsigned)(BMT * LDT * 2);
#pragma unroll
        for (int r = 0; r < 4; ++r) {
            int ch = tid + r * 256;
            int row = ch >> 3, cc = ch & 7;
            int eo = row * LDT + cc * 8;
            g2l_chunk(xg + eo, xb + (unsigned)(eo * 2), A0,
                      (row * CR + k0 + cc * 8) * 2);
        }
#pragma unroll
        for (int r = 0; r < 2; ++r) {
            int ch = tid + r * 256;
            int row = ch >> 3, cc = ch & 7;
            int eo = row * LDT + cc * 8;
            g2l_chunk(wg + eo, wb + (unsigned)(eo * 2), B0,
                      (row * CR + k0 + cc * 8) * 2);
        }
    };
    auto compute = [&](int buf) {
        const unsigned short* Xb = smem + buf * (BMT + BNT) * LDT;
        const unsigned short* Wb = Xb + BMT * LDT;
#pragma unroll
        for (int ks = 0; ks < BKT; ks += 32) {
            v16bf a0f = load_frag_row(&Xb[(wm + laneLo) * LDT + ks], laneHi);
            v16bf a1f = load_frag_row(&Xb[(wm + 16 + laneLo) * LDT + ks], laneHi);
            v16bf b0f = load_frag_row(&Wb[(wn + laneLo) * LDT + ks], laneHi);
            v16bf b1f = load_frag_row(&Wb[(wn + 16 + laneLo) * LDT + ks], laneHi);
            acc[0][0] = wmma_bf16(a0f, b0f, acc[0][0]);
            acc[0][1] = wmma_bf16(a0f, b1f, acc[0][1]);
            acc[1][0] = wmma_bf16(a1f, b0f, acc[1][0]);
            acc[1][1] = wmma_bf16(a1f, b1f, acc[1][1]);
        }
    };

    issue(0, 0);
    wait_g2l();
    __syncthreads();
    int buf = 0;
    const int NSTAGE = NADP * (CR / BKT);   // 40
    for (int s = 0; s < NSTAGE; ++s) {
        if (s + 1 < NSTAGE) issue(s + 1, buf ^ 1);
        compute(buf);
        if ((s & 3) == 3) {                 // adapter boundary: fold with cnt
            const int aa = s >> 2;
#pragma unroll
            for (int ti = 0; ti < 2; ++ti) {
#pragma unroll
                for (int i = 0; i < 8; ++i) {
                    float cw = cntS[wm + ti * 16 + laneHi * 8 + i][aa];
#pragma unroll
                    for (int tj = 0; tj < 2; ++tj)
                        tot[ti][tj][i] += cw * acc[ti][tj][i];
                }
            }
            acc[0][0] = (v8f){}; acc[0][1] = (v8f){};
            acc[1][0] = (v8f){}; acc[1][1] = (v8f){};
        }
        wait_g2l();
        __syncthreads();
        buf ^= 1;
    }

#pragma unroll
    for (int ti = 0; ti < 2; ++ti) {
#pragma unroll
        for (int tj = 0; tj < 2; ++tj) {
            const int c = c0 + wn + tj * 16 + laneLo;
#pragma unroll
            for (int i = 0; i < 8; ++i) {
                int tok = t0 + wm + ti * 16 + laneHi * 8 + i;
                int b = tok >> 12, hw = tok & (HWSZ - 1);
                size_t o = ((size_t)b * CCH + c) * HWSZ + hw;
                out[o] = tot[ti][tj][i] + feat[o];
            }
        }
    }
}

// ---------------- launcher ----------------

extern "C" void kernel_launch(void* const* d_in, const int* in_sizes, int n_in,
                              void* d_out, int out_size, void* d_ws, size_t ws_size,
                              hipStream_t stream) {
    const float* feat = (const float*)d_in[0];   // [B, C, H, W]
    const float* ar   = (const float*)d_in[1];   // [B, NA, H, W]
    const float* lnsc = (const float*)d_in[2];   // [A, C]
    const float* lnb  = (const float*)d_in[3];   // [A, C]
    const float* w1   = (const float*)d_in[4];   // [A, C, Cr]
    const float* w2   = (const float*)d_in[5];   // [A, Cr, C]
    float* out = (float*)d_out;

    char* ws = (char*)d_ws;
    size_t off = 0;
    auto alloc = [&](size_t bytes) -> void* {
        void* p = ws + off;
        off = (off + bytes + 255) & ~(size_t)255;
        return p;
    };
    unsigned short* xn   = (unsigned short*)alloc((size_t)NTOK * CCH * 2);       // 16.8 MB
    unsigned short* hbuf = (unsigned short*)alloc((size_t)NADP * NTOK * CR * 2); // 41.9 MB
    unsigned short* w1t  = (unsigned short*)alloc((size_t)NADP * CR * CCH * 2);  // 5.2 MB
    unsigned short* w2t  = (unsigned short*)alloc((size_t)NADP * CCH * CR * 2);  // 5.2 MB
    float* hb  = (float*)alloc((size_t)NADP * CR * 4);
    float* cnt = (float*)alloc((size_t)NTOK * NADP * 4);

    saa_prep_w1<<<1024, 256, 0, stream>>>(w1, lnsc, w1t);
    saa_prep_w2<<<1024, 256, 0, stream>>>(w2, w2t);
    saa_prep_hbias<<<NADP, CR, 0, stream>>>(w1, lnb, hb);
    saa_ln<<<NTOK / 32, 256, 1024 * 33 * sizeof(float), stream>>>(feat, xn);
    saa_cnt<<<NTOK / 256, 256, 0, stream>>>(ar, cnt);
    saa_gemm1<<<dim3(NTOK / BMT, CR / BNT, NADP), 256, 0, stream>>>(xn, w1t, hb, hbuf);
    saa_gemm2<<<dim3(NTOK / BMT, CCH / BNT), 256, 0, stream>>>(hbuf, w2t, cnt, feat, out);
}
